// DescSelector_41446434406632
// MI455X (gfx1250) — matrix operand order repaired and checked by reading, not compiled, not run on hardware
//
#include <hip/hip_runtime.h>

typedef __attribute__((ext_vector_type(16))) _Float16 v16h;
typedef __attribute__((ext_vector_type(8)))  _Float16 v8h;
typedef __attribute__((ext_vector_type(8)))  float    v8f;

#define IN_DIM   256
#define HID_DIM  256
#define OUT_DIM  128
#define NUM_BINS 128
#define LN_EPS   1e-5f

union H2U { _Float16 h[2]; unsigned u; };

// ---------------------------------------------------------------------------
// CDNA5 async global->LDS copy (16B per lane), tracked by ASYNCcnt.
// LDS operand: AS3 byte offset == low 32 bits of the generic shared pointer.
// ---------------------------------------------------------------------------
__device__ __forceinline__ void async_copy_b128(void* ldsDst, const void* gsrc) {
  unsigned ldsOff = (unsigned)(unsigned long long)ldsDst;
  asm volatile("global_load_async_to_lds_b128 %0, %1, off"
               :: "v"(ldsOff), "v"(gsrc) : "memory");
}
__device__ __forceinline__ void wait_async0() {
  asm volatile("s_wait_asynccnt 0x0" ::: "memory");
}

// ---------------------------------------------------------------------------
// Repack W1 / W2 into CDNA5 WMMA B-fragment layout:
//   WB: [nt][ks=8][lane=32][v=8] u32, value = pack(W[k0][n], W[k0+1][n])
//       n = nt*16 + (lane&15), k0 = ks*32 + (lane>>4)*8 + 2v + (v>=4 ? 8 : 0)
// so each lane reads its 8-VGPR B fragment as one contiguous 32B chunk.
// ---------------------------------------------------------------------------
__global__ void prep_weights(const float* __restrict__ W1,
                             const float* __restrict__ W2,
                             unsigned* __restrict__ W1B,
                             unsigned* __restrict__ W2B) {
  int i = blockIdx.x * blockDim.x + threadIdx.x;
  const int totW1 = 16 * 8 * 32 * 8;   // 32768
  const int totW2 = 8 * 8 * 32 * 8;    // 16384
  if (i < totW1) {
    int v = i & 7, lane = (i >> 3) & 31, ks = (i >> 8) & 7, nt = i >> 11;
    int n  = nt * 16 + (lane & 15);
    int k0 = ks * 32 + (lane >> 4) * 8 + 2 * v + (v >= 4 ? 8 : 0);
    H2U p; p.h[0] = (_Float16)W1[k0 * HID_DIM + n];
           p.h[1] = (_Float16)W1[(k0 + 1) * HID_DIM + n];
    W1B[i] = p.u;
  } else if (i < totW1 + totW2) {
    int j = i - totW1;
    int v = j & 7, lane = (j >> 3) & 31, ks = (j >> 8) & 7, nt = j >> 11;
    int n  = nt * 16 + (lane & 15);
    int k0 = ks * 32 + (lane >> 4) * 8 + 2 * v + (v >= 4 ? 8 : 0);
    H2U p; p.h[0] = (_Float16)W2[k0 * OUT_DIM + n];
           p.h[1] = (_Float16)W2[(k0 + 1) * OUT_DIM + n];
    W2B[j] = p.u;
  }
}

__global__ void zero_keys(unsigned long long* __restrict__ keys, int n) {
  int i = blockIdx.x * blockDim.x + threadIdx.x;
  if (i < n) keys[i] = 0ull;
}

// ---------------------------------------------------------------------------
// Fused MLP, 8 waves per block, 16 rows per wave (M = 128 rows / block).
// Weight tiles staged once per block via async global->LDS DMA, double
// buffered so the DMA overlaps the WMMA chain. Hidden layer lives entirely
// in registers; LayerNorm via wave32 half-shuffle reductions.
// ---------------------------------------------------------------------------
__global__ __launch_bounds__(256)
void mlp_scores(const float* __restrict__ desc,
                const float* __restrict__ b1, const float* __restrict__ g1,
                const float* __restrict__ be1, const float* __restrict__ b2,
                const float* __restrict__ Ws, const float* __restrict__ bs,
                const unsigned* __restrict__ W1B,
                const unsigned* __restrict__ W2B,
                float* __restrict__ scores) {
  __shared__ alignas(16) unsigned ldsB[2][8 * 32 * 8];   // 2 x 8 KB weight stage
  __shared__ alignas(16) _Float16 ldsN[8][16 * 256];     // 8 x 8 KB activations

  const int tid  = threadIdx.x;
  const int wave = tid >> 5;
  const int lane = tid & 31;
  const int lo = lane & 15, hi = lane >> 4;
  const long long rowbase = (long long)blockIdx.x * 128 + wave * 16;
  const float* Xrow = desc + (rowbase + lo) * (long long)IN_DIM;

  // warm W2B into L2 for later (global_prefetch_b8)
  __builtin_prefetch(W2B + (tid << 6), 0, 1);

  // per-thread async stage of 32B of weight tile nt into ldsB[buf]
  auto stageAsync = [&](const unsigned* __restrict__ WB, int nt, int buf) {
    const unsigned* src = WB + (((nt * 8 + wave) * 32 + lane) << 3);
    unsigned* dst = &ldsB[buf][(wave * 32 + lane) << 3];
    async_copy_b128(dst, src);
    async_copy_b128(dst + 4, src + 4);
  };

  // ---- GEMM1 A fragments straight from global desc (f32 -> f16) ----------
  // 16-bit 16x32 A layout (ISA 7.12.2): lane half hi covers K = ks*32+hi*8
  // + {0..7} and + {16..23}; the two halves read disjoint K ranges.
  v16h afrag[8];
#pragma unroll
  for (int ks = 0; ks < 8; ++ks) {
    const float4* p = (const float4*)(Xrow + ks * 32 + hi * 8);
    const float4* q = (const float4*)(Xrow + ks * 32 + hi * 8 + 16);
    float4 g0 = p[0], g1v = p[1], g2 = q[0], g3 = q[1];
    v16h a;
    a[0]  = (_Float16)g0.x;  a[1]  = (_Float16)g0.y;
    a[2]  = (_Float16)g0.z;  a[3]  = (_Float16)g0.w;
    a[4]  = (_Float16)g1v.x; a[5]  = (_Float16)g1v.y;
    a[6]  = (_Float16)g1v.z; a[7]  = (_Float16)g1v.w;
    a[8]  = (_Float16)g2.x;  a[9]  = (_Float16)g2.y;
    a[10] = (_Float16)g2.z;  a[11] = (_Float16)g2.w;
    a[12] = (_Float16)g3.x;  a[13] = (_Float16)g3.y;
    a[14] = (_Float16)g3.z;  a[15] = (_Float16)g3.w;
    afrag[ks] = a;
  }

  // ---- GEMM1: H = X @ W1  (16 n-tiles x 8 k-steps, all acc live) ---------
  v8f acc[16];
  stageAsync(W1B, 0, 0);
  wait_async0();
  __syncthreads();
  int buf = 0;
#pragma unroll
  for (int nt = 0; nt < 16; ++nt) {
    if (nt + 1 < 16) stageAsync(W1B, nt + 1, buf ^ 1);  // DMA overlaps WMMAs
    v16h bf[8];
#pragma unroll
    for (int ks = 0; ks < 8; ++ks)                      // one clause of loads
      bf[ks] = *(const v16h*)&ldsB[buf][(ks * 32 + lane) << 3];
    v8f c = {};
#pragma unroll
    for (int ks = 0; ks < 8; ++ks)                      // back-to-back WMMAs
      c = __builtin_amdgcn_wmma_f32_16x16x32_f16(
          false, afrag[ks], false, bf[ks], (short)0, c, false, false);
    acc[nt] = c;
    if (nt + 1 < 16) {
      wait_async0();
      __syncthreads();
      buf ^= 1;
    }
  }

  // ---- bias + LayerNorm stats in registers (exact f32) -------------------
  float sum[8], sq[8];
#pragma unroll
  for (int r = 0; r < 8; ++r) { sum[r] = 0.f; sq[r] = 0.f; }
#pragma unroll
  for (int nt = 0; nt < 16; ++nt) {
    float bb = b1[nt * 16 + lo];
#pragma unroll
    for (int r = 0; r < 8; ++r) {
      float h = acc[nt][r] + bb;
      acc[nt][r] = h;
      sum[r] += h; sq[r] += h * h;
    }
  }
#pragma unroll
  for (int m = 1; m < 16; m <<= 1) {      // reduce over the 16-lane half
#pragma unroll
    for (int r = 0; r < 8; ++r) {
      sum[r] += __shfl_xor(sum[r], m, 32);
      sq[r]  += __shfl_xor(sq[r],  m, 32);
    }
  }
  float mu[8], inv[8];
#pragma unroll
  for (int r = 0; r < 8; ++r) {
    mu[r] = sum[r] * (1.f / HID_DIM);
    float var = sq[r] * (1.f / HID_DIM) - mu[r] * mu[r];
    inv[r] = rsqrtf(var + LN_EPS);
  }

  // ---- normalize + SiLU -> f16 LDS slice (A operand for GEMM2) -----------
  _Float16* Nw = ldsN[wave];
#pragma unroll
  for (int nt = 0; nt < 16; ++nt) {
    int col = nt * 16 + lo;
    float g = g1[col], be = be1[col];
#pragma unroll
    for (int r = 0; r < 8; ++r) {
      float x = (acc[nt][r] - mu[r]) * inv[r] * g + be;
      x = x / (1.f + __expf(-x));                       // SiLU
      Nw[(r + hi * 8) * HID_DIM + col] = (_Float16)x;
    }
  }
  __syncthreads();   // activations visible; all GEMM1 ldsB reads done

  // stage first W2B tile now: DMA overlaps the A-fragment rebuild below
  stageAsync(W2B, 0, 0);

  // ---- GEMM2 A fragments from normalized activations ---------------------
#pragma unroll
  for (int ks = 0; ks < 8; ++ks) {
    const v8h a0 = *(const v8h*)&Nw[lo * HID_DIM + ks * 32 + hi * 8];
    const v8h a1 = *(const v8h*)&Nw[lo * HID_DIM + ks * 32 + hi * 8 + 16];
    v16h a;
#pragma unroll
    for (int j = 0; j < 8; ++j) { a[j] = a0[j]; a[j + 8] = a1[j]; }
    afrag[ks] = a;
  }
  wait_async0();
  __syncthreads();

  // ---- GEMM2: F = Hn @ W2  (8 n-tiles x 8 k-steps) -----------------------
  v8f acc2[8];
  buf = 0;
#pragma unroll
  for (int nt = 0; nt < 8; ++nt) {
    if (nt + 1 < 8) stageAsync(W2B, nt + 1, buf ^ 1);
    v16h bf[8];
#pragma unroll
    for (int ks = 0; ks < 8; ++ks)
      bf[ks] = *(const v16h*)&ldsB[buf][(ks * 32 + lane) << 3];
    v8f c = {};
#pragma unroll
    for (int ks = 0; ks < 8; ++ks)
      c = __builtin_amdgcn_wmma_f32_16x16x32_f16(
          false, afrag[ks], false, bf[ks], (short)0, c, false, false);
    acc2[nt] = c;
    if (nt + 1 < 8) {
      wait_async0();
      __syncthreads();
      buf ^= 1;
    }
  }

  // ---- score head in registers: score = (F + b2) . Ws + bs ---------------
  float sc[8];
#pragma unroll
  for (int r = 0; r < 8; ++r) sc[r] = 0.f;
#pragma unroll
  for (int nt = 0; nt < 8; ++nt) {
    int col = nt * 16 + lo;
    float w = Ws[col], bb = b2[col];
#pragma unroll
    for (int r = 0; r < 8; ++r) sc[r] += (acc2[nt][r] + bb) * w;
  }
#pragma unroll
  for (int m = 1; m < 16; m <<= 1)
#pragma unroll
    for (int r = 0; r < 8; ++r) sc[r] += __shfl_xor(sc[r], m, 32);
  if (lo == 0) {
    float bss = bs[0];
#pragma unroll
    for (int r = 0; r < 8; ++r)
      scores[rowbase + hi * 8 + r] = sc[r] + bss;
  }
}

// ---------------------------------------------------------------------------
// Grid-id + per-bin argmax via packed u64 atomicMax.
// Key = monotone(score) << 32 | (0x7FFFFFFF - idx): max score, min idx on tie.
// ---------------------------------------------------------------------------
__global__ void bin_kernel(const float* __restrict__ kpts,
                           const float* __restrict__ scores,
                           const int* __restrict__ pH, const int* __restrict__ pW,
                           unsigned long long* __restrict__ keys,
                           int* __restrict__ flags,
                           long long BN, int N) {
  long long i = (long long)blockIdx.x * blockDim.x + threadIdx.x;
  if (i >= BN) return;
  flags[i] = 0;
  float x = kpts[2 * i], y = kpts[2 * i + 1];
  float Hf = (float)pH[0], Wf = (float)pW[0];
  bool mid = (y > 0.2f * Hf) && (y <= 0.5f * Hf);
  bool btm = (y > 0.5f * Hf);
  int gid = -1;
  if (btm) {
    int bgx = (int)(x / Wf * 16.f);                        bgx = min(max(bgx, 0), 15);
    int bgy = (int)((y - 0.5f * Hf) / (0.5f * Hf) * 6.f);  bgy = min(max(bgy, 0), 5);
    gid = 32 + bgy * 16 + bgx;
  } else if (mid) {
    int mgx = (int)(x / Wf * 8.f);                         mgx = min(max(mgx, 0), 7);
    int mgy = (int)((y - 0.2f * Hf) / (0.3f * Hf) * 4.f);  mgy = min(max(mgy, 0), 3);
    gid = mgy * 8 + mgx;
  }
  if (gid >= 0) {
    int b = (int)(i / N), n = (int)(i % N);
    unsigned u = __float_as_uint(scores[i]);
    u = (u & 0x80000000u) ? ~u : (u | 0x80000000u);   // monotone order map
    unsigned long long key =
        ((unsigned long long)u << 32) | (unsigned)(0x7FFFFFFF - n);
    atomicMax(&keys[b * NUM_BINS + gid], key);
  }
}

// ---------------------------------------------------------------------------
// Per-batch: compact bin winners (bin order) + iterative masked-argmax top-k
// fallback (matches jax.lax.top_k tie order: lowest index first).
// ---------------------------------------------------------------------------
__global__ void select_kernel(const float* __restrict__ scores,
                              const unsigned long long* __restrict__ keys,
                              int* __restrict__ flags, int* __restrict__ sel,
                              const int* __restrict__ ptopk, int N) {
  __shared__ int   s_sel[256];
  __shared__ int   s_num;
  __shared__ float s_bv[256];
  __shared__ int   s_bi[256];
  int b = blockIdx.x;
  int topk = ptopk[0]; if (topk > 256) topk = 256;
  const unsigned long long* kb = keys + b * NUM_BINS;
  const float* sc = scores + (long long)b * N;
  int* fl = flags + (long long)b * N;

  if (threadIdx.x == 0) {
    int cnt = 0;
    for (int g = 0; g < NUM_BINS; ++g) {
      unsigned long long k = kb[g];
      if (k) s_sel[cnt++] = 0x7FFFFFFF - (int)(k & 0xFFFFFFFFu);
    }
    s_num = cnt;
  }
  __syncthreads();
  int num = s_num;
  for (int j = threadIdx.x; j < num; j += blockDim.x) fl[s_sel[j]] = 1;
  __syncthreads();

  for (int r = num; r < topk; ++r) {
    float bv = -3.4e38f; int bi = N;
    for (int i = threadIdx.x; i < N; i += (int)blockDim.x)
      if (!fl[i]) { float v = sc[i]; if (v > bv) { bv = v; bi = i; } }
    s_bv[threadIdx.x] = bv; s_bi[threadIdx.x] = bi;
    __syncthreads();
    if (threadIdx.x == 0) {
      float BV = -3.4e38f; int BI = N;
      for (int t = 0; t < (int)blockDim.x; ++t) {
        float v = s_bv[t]; int i2 = s_bi[t];
        if (v > BV || (v == BV && i2 < BI)) { BV = v; BI = i2; }
      }
      s_sel[r] = BI; if (BI < N) fl[BI] = 1;
    }
    __syncthreads();
  }
  for (int j = threadIdx.x; j < topk; j += blockDim.x) sel[b * 256 + j] = s_sel[j];
}

// ---------------------------------------------------------------------------
// Gather outputs: [B,topk,256] feat | [B,topk,2] kpts | [B,topk] indices(float)
// ---------------------------------------------------------------------------
__global__ void gather_kernel(const float* __restrict__ desc,
                              const float* __restrict__ kpts,
                              const int* __restrict__ sel,
                              const int* __restrict__ ptopk,
                              float* __restrict__ out, int N, int B) {
  int topk = ptopk[0]; if (topk > 256) topk = 256;
  int b = blockIdx.x >> 8, j = blockIdx.x & 255;
  if (j >= topk) return;
  int idx = sel[b * 256 + j];
  long long slot = (long long)b * topk + j;
  const float* srow = desc + ((long long)b * N + idx) * IN_DIM;
  float* frow = out + slot * IN_DIM;
  for (int t = threadIdx.x; t < IN_DIM; t += (int)blockDim.x) frow[t] = srow[t];
  if (threadIdx.x == 0) {
    long long kbase = (long long)B * topk * IN_DIM;
    out[kbase + slot * 2]     = kpts[((long long)b * N + idx) * 2];
    out[kbase + slot * 2 + 1] = kpts[((long long)b * N + idx) * 2 + 1];
    out[kbase + (long long)B * topk * 2 + slot] = (float)idx;
  }
}

extern "C" void kernel_launch(void* const* d_in, const int* in_sizes, int n_in,
                              void* d_out, int out_size, void* d_ws, size_t ws_size,
                              hipStream_t stream) {
  const float* kpts = (const float*)d_in[0];
  const float* desc = (const float*)d_in[1];
  const int*   pH   = (const int*)d_in[2];
  const int*   pW   = (const int*)d_in[3];
  const int*   ptk  = (const int*)d_in[4];
  const float* W1   = (const float*)d_in[5];
  const float* b1   = (const float*)d_in[6];
  const float* g1   = (const float*)d_in[7];
  const float* be1  = (const float*)d_in[8];
  const float* W2   = (const float*)d_in[9];
  const float* b2   = (const float*)d_in[10];
  const float* Ws   = (const float*)d_in[11];
  const float* bs   = (const float*)d_in[12];

  const long long BN = (long long)in_sizes[1] / IN_DIM;  // B*N
  const int B = 8;                                        // per reference setup
  const int N = (int)(BN / B);

  // workspace carve-up (256B aligned)
  char* w = (char*)d_ws;
  auto carve = [&](size_t bytes) { char* p = w; w += (bytes + 255) & ~(size_t)255; return p; };
  float*              scores = (float*)carve((size_t)BN * 4);
  unsigned long long* keys   = (unsigned long long*)carve((size_t)B * NUM_BINS * 8);
  int*                flags  = (int*)carve((size_t)BN * 4);
  int*                sel    = (int*)carve((size_t)B * 256 * 4);
  unsigned*           W1B    = (unsigned*)carve((size_t)16 * 8 * 32 * 8 * 4);
  unsigned*           W2B    = (unsigned*)carve((size_t)8 * 8 * 32 * 8 * 4);

  const int totPack = 16 * 8 * 32 * 8 + 8 * 8 * 32 * 8;
  prep_weights<<<(totPack + 255) / 256, 256, 0, stream>>>(W1, W2, W1B, W2B);
  zero_keys<<<(B * NUM_BINS + 255) / 256, 256, 0, stream>>>(keys, B * NUM_BINS);

  mlp_scores<<<(int)(BN / 128), 256, 0, stream>>>(desc, b1, g1, be1, b2, Ws, bs,
                                                  W1B, W2B, scores);
  bin_kernel<<<(int)((BN + 255) / 256), 256, 0, stream>>>(kpts, scores, pH, pW,
                                                          keys, flags, BN, N);
  select_kernel<<<B, 256, 0, stream>>>(scores, keys, flags, sel, ptk, N);
  gather_kernel<<<B * 256, 64, 0, stream>>>(desc, kpts, sel, ptk,
                                            (float*)d_out, N, B);
}
